// GMReader2GATSortPoolReadout_15178414424428
// MI455X (gfx1250) — matrix operand order repaired and checked by reading, not compiled, not run on hardware
//
#include <hip/hip_runtime.h>
#include <hip/hip_bf16.h>

// ---------------- problem constants (from reference) ----------------
constexpr int NTOT  = 32768;    // nodes
constexpr int BGR   = 32;       // graphs
constexpr int NPERG = 1024;     // nodes per graph
constexpr int NEDGE = 524288;   // edges
constexpr int KTOP  = 32;       // sortpool top-k
constexpr int DHID  = 64;       // hidden dim per head
constexpr float NEG_ATTN = 0.2f;
constexpr float NEG_ACT  = 0.01f;

typedef __attribute__((ext_vector_type(16))) _Float16 v16h;
typedef __attribute__((ext_vector_type(8)))  float    v8f;

// ---------------- WMMA GEMM: C[M,P] = A[M,K] * B[K,P] (+bias, optional relu)
// Row-major fp32 in/out, f16 multiply with f32 accumulate via
// v_wmma_f32_16x16x32_f16. One wave computes a 16 x (16*NT) tile: A loaded
// once per K-step (as 4x float4 = global_load_b128) and shared by NT WMMAs.
template <int NT>
__global__ __launch_bounds__(256)
void k_wmma_gemm(const float* __restrict__ A, const float* __restrict__ Bw,
                 const float* __restrict__ bias, float* __restrict__ C,
                 int M, int Kd, int P, int relu)
{
  const int lane = threadIdx.x & 31;
  const int wave = (blockIdx.x * (blockDim.x >> 5)) + (threadIdx.x >> 5);
  const int groupsP = P / (16 * NT);
  const int totalTiles = (M >> 4) * groupsP;
  if (wave >= totalTiles) return;              // wave-uniform: EXEC stays all-ones
  const int m0 = (wave / groupsP) << 4;
  const int p0 = (wave % groupsP) * (16 * NT);

  const int mrow = lane & 15;                  // A row within tile / B,C column
  const int g    = lane >> 4;                  // lane-group select (ISA layout)

  const v8f vzero = {};
  v8f acc[NT];
#pragma unroll
  for (int t = 0; t < NT; ++t) acc[t] = vzero;

  for (int k0 = 0; k0 < Kd; k0 += 32) {
    const float* Ar = A + (size_t)(m0 + mrow) * Kd + k0;
    // prefetch next K-step operands (-> global_prefetch_b8)
    if (k0 + 32 < Kd) {
      __builtin_prefetch(Ar + 32, 0, 3);
      __builtin_prefetch(Bw + (size_t)(k0 + 32) * P + p0 + mrow, 0, 3);
    }
    // A tile 16x32 (ISA layout): lane spans k = g*8..+7 and 16+g*8..+7,
    // both 16B-aligned -> four b128 loads, then pack to f16.
    float4 alo0 = *(const float4*)(Ar + g * 8);
    float4 alo1 = *(const float4*)(Ar + g * 8 + 4);
    float4 ahi0 = *(const float4*)(Ar + 16 + g * 8);
    float4 ahi1 = *(const float4*)(Ar + 16 + g * 8 + 4);
    v16h a;
    a[0]  = (_Float16)alo0.x; a[1]  = (_Float16)alo0.y;
    a[2]  = (_Float16)alo0.z; a[3]  = (_Float16)alo0.w;
    a[4]  = (_Float16)alo1.x; a[5]  = (_Float16)alo1.y;
    a[6]  = (_Float16)alo1.z; a[7]  = (_Float16)alo1.w;
    a[8]  = (_Float16)ahi0.x; a[9]  = (_Float16)ahi0.y;
    a[10] = (_Float16)ahi0.z; a[11] = (_Float16)ahi0.w;
    a[12] = (_Float16)ahi1.x; a[13] = (_Float16)ahi1.y;
    a[14] = (_Float16)ahi1.z; a[15] = (_Float16)ahi1.w;

#pragma unroll
    for (int t = 0; t < NT; ++t) {
      // B tile 32x16 (ISA layout, matches A): lane spans k = g*8..+7 and 16+g*8..+7; 2 K/VGPR
      const float* Br = Bw + (size_t)k0 * P + p0 + t * 16 + mrow;
      v16h b;
#pragma unroll
      for (int v = 0; v < 8; ++v) {
        int kk = g * 8 + 2 * v + (v >= 4 ? 8 : 0);
        b[2 * v]     = (_Float16)Br[(size_t)kk * P];
        b[2 * v + 1] = (_Float16)Br[(size_t)(kk + 1) * P];
      }
      acc[t] = __builtin_amdgcn_wmma_f32_16x16x32_f16(false, a, false, b,
                                                      (short)0, acc[t], false, false);
    }
  }
  // C/D layout: VGPR v -> row v + 8*g, column = lane&15
#pragma unroll
  for (int t = 0; t < NT; ++t) {
    const int pc = p0 + t * 16 + mrow;
    const float bv = bias ? bias[pc] : 0.f;
#pragma unroll
    for (int v = 0; v < 8; ++v) {
      int mr = m0 + v + (g << 3);
      float val = acc[t][v] + bv;
      if (relu) val = fmaxf(val, 0.f);
      C[(size_t)mr * P + pc] = val;
    }
  }
}

// ---------------- helpers ----------------
__global__ void k_fill(float* p, float v, size_t n) {
  size_t i = (size_t)blockIdx.x * blockDim.x + threadIdx.x;
  if (i < n) p[i] = v;
}

__device__ inline void atomicMaxF(float* addr, float val) {
  unsigned int* ua = (unsigned int*)addr;
  unsigned int old = *ua;
  while (__uint_as_float(old) < val) {
    unsigned int assumed = old;
    old = atomicCAS(ua, assumed, __float_as_uint(val));
    if (old == assumed) break;
  }
}

// el[n,h] = f[n,h,:] . al[h,:]  ;  er likewise
__global__ void k_attn_scores(const float* __restrict__ f, const float* __restrict__ al,
                              const float* __restrict__ ar, float* __restrict__ el,
                              float* __restrict__ er, int H)
{
  int i = blockIdx.x * blockDim.x + threadIdx.x;  // i over N*H
  if (i >= NTOT * H) return;
  int h = i % H;
  const float* fp  = f  + (size_t)i * DHID;
  const float* alp = al + (size_t)h * DHID;
  const float* arp = ar + (size_t)h * DHID;
  float sl = 0.f, sr = 0.f;
#pragma unroll 8
  for (int d = 0; d < DHID; ++d) { float v = fp[d]; sl += v * alp[d]; sr += v * arp[d]; }
  el[i] = sl; er[i] = sr;
}

// pass 1: e = leakyrelu(el[src]+er[dst], 0.2); running max per (dst,h)
__global__ void k_edge_score(const int* __restrict__ src, const int* __restrict__ dst,
                             const float* __restrict__ el, const float* __restrict__ er,
                             float* __restrict__ ebuf, float* __restrict__ m, int H)
{
  long long idx = (long long)blockIdx.x * blockDim.x + threadIdx.x;
  if (idx >= (long long)NEDGE * H) return;
  int e = (int)(idx / H), h = (int)(idx % H);
  int s = src[e], d = dst[e];
  float x = el[s * H + h] + er[d * H + h];
  x = (x > 0.f) ? x : NEG_ATTN * x;
  ebuf[idx] = x;
  atomicMaxF(&m[d * H + h], x);
}

// pass 2: p = exp(e - m[dst]); z[dst] += p   (m guarded like ref's isfinite)
__global__ void k_edge_exp(const int* __restrict__ dst, float* __restrict__ ebuf,
                           const float* __restrict__ m, float* __restrict__ z, int H)
{
  long long idx = (long long)blockIdx.x * blockDim.x + threadIdx.x;
  if (idx >= (long long)NEDGE * H) return;
  int e = (int)(idx / H), h = (int)(idx % H);
  int d = dst[e];
  float mm = m[d * H + h];
  if (!(mm > -3.0e38f && mm < 3.0e38f)) mm = 0.f;
  float p = __expf(ebuf[idx] - mm);
  ebuf[idx] = p;
  atomicAdd(&z[d * H + h], p);
}

// pass 3: hout[dst,h,:] += (p / max(z,1e-9)) * f[src,h,:]
__global__ void k_edge_aggr(const int* __restrict__ src, const int* __restrict__ dst,
                            const float* __restrict__ pbuf, const float* __restrict__ z,
                            const float* __restrict__ f, float* __restrict__ hout, int H)
{
  long long idx = (long long)blockIdx.x * blockDim.x + threadIdx.x;
  if (idx >= (long long)NEDGE * H) return;
  int e = (int)(idx / H), h = (int)(idx % H);
  int s = src[e], d = dst[e];
  float a = pbuf[idx] / fmaxf(z[d * H + h], 1e-9f);
  const float* fs = f    + ((size_t)s * H + h) * DHID;
  float*       ho = hout + ((size_t)d * H + h) * DHID;
#pragma unroll 8
  for (int dd = 0; dd < DHID; ++dd) atomicAdd(&ho[dd], a * fs[dd]);
}

// GraphNorm (per graph, per head, per channel) + LeakyReLU(0.01), in place.
// One block per (graph b, head h): 256 threads = 4 node-groups x 64 channels.
__global__ __launch_bounds__(256)
void k_graphnorm_lrelu(float* __restrict__ x, const float* __restrict__ alpha,
                       const float* __restrict__ gamma, const float* __restrict__ beta,
                       int H)
{
  __shared__ float s1[256], s2[256];
  const int b = blockIdx.x / H, h = blockIdx.x % H;
  const int t = threadIdx.x, d = t & 63, grp = t >> 6;  // grp in 0..3
  float sum = 0.f, sumsq = 0.f;
  for (int n = grp * 256; n < (grp + 1) * 256; ++n) {
    float v = x[(((size_t)b * NPERG + n) * H + h) * DHID + d];
    sum += v; sumsq += v * v;
  }
  s1[t] = sum; s2[t] = sumsq;
  __syncthreads();
  if (grp == 0) {
    for (int g2 = 1; g2 < 4; ++g2) { sum += s1[g2 * 64 + d]; sumsq += s2[g2 * 64 + d]; }
    s1[d] = sum; s2[d] = sumsq;
  }
  __syncthreads();
  const float S = s1[d], Q = s2[d];
  const float mean = S * (1.f / NPERG);
  const float a = alpha[d];
  const float var = Q * (1.f / NPERG) - 2.f * a * mean * mean + a * a * mean * mean;
  const float inv = rsqrtf(var + 1e-5f);
  const float gm = gamma[d], be = beta[d];
  for (int n = grp; n < NPERG; n += 4) {
    size_t off = (((size_t)b * NPERG + n) * H + h) * DHID + d;
    float y = gm * (x[off] - a * mean) * inv + be;
    x[off] = (y > 0.f) ? y : NEG_ACT * y;
  }
}

// xm[n,d] = mean over heads of x[n,h,d]
__global__ void k_headmean(const float* __restrict__ x, float* __restrict__ xm, int H)
{
  int i = blockIdx.x * blockDim.x + threadIdx.x;   // over N*64
  if (i >= NTOT * DHID) return;
  int n = i >> 6, d = i & 63;
  float s = 0.f;
  for (int h = 0; h < H; ++h) s += x[((size_t)n * H + h) * DHID + d];
  xm[i] = s * (1.f / H);
}

// per-node ascending sort of 64 features; key = max (last sorted channel)
__global__ void k_sortnode(const float* __restrict__ xm, float* __restrict__ xs,
                           float* __restrict__ key)
{
  int n = blockIdx.x * blockDim.x + threadIdx.x;
  if (n >= NTOT) return;
  float v[DHID];
#pragma unroll
  for (int i = 0; i < DHID; ++i) v[i] = xm[(size_t)n * DHID + i];
  for (int i = 1; i < DHID; ++i) {        // insertion sort ascending
    float xv = v[i]; int j = i - 1;
    while (j >= 0 && v[j] > xv) { v[j + 1] = v[j]; --j; }
    v[j + 1] = xv;
  }
#pragma unroll
  for (int i = 0; i < DHID; ++i) xs[(size_t)n * DHID + i] = v[i];
  key[n] = v[DHID - 1];
}

// top-K selection per graph (descending, first-index tie-break) + gather rows
__global__ __launch_bounds__(256)
void k_toppool(const float* __restrict__ xs, const float* __restrict__ key,
               float* __restrict__ r, int roff)
{
  __shared__ float sk[NPERG];
  __shared__ int   chosen[KTOP];
  __shared__ float rv[256];
  __shared__ int   ri[256];
  const int b = blockIdx.x, t = threadIdx.x;
  for (int i = t; i < NPERG; i += 256) sk[i] = key[b * NPERG + i];
  __syncthreads();
  for (int it = 0; it < KTOP; ++it) {
    float bv = -3.4e38f; int bi = NPERG;
    for (int i = t; i < NPERG; i += 256) {
      float v = sk[i];
      if (v > bv || (v == bv && i < bi)) { bv = v; bi = i; }
    }
    rv[t] = bv; ri[t] = bi;
    __syncthreads();
    for (int s = 128; s > 0; s >>= 1) {
      if (t < s) {
        if (rv[t + s] > rv[t] || (rv[t + s] == rv[t] && ri[t + s] < ri[t])) {
          rv[t] = rv[t + s]; ri[t] = ri[t + s];
        }
      }
      __syncthreads();
    }
    if (t == 0) { chosen[it] = ri[0]; sk[ri[0]] = -3.4e38f; }
    __syncthreads();
  }
  for (int j = t; j < KTOP * DHID; j += 256) {
    int node = chosen[j >> 6], d = j & 63;
    r[(size_t)b * (2 * KTOP * DHID) + roff + j] =
        xs[((size_t)b * NPERG + node) * DHID + d];
  }
}

// ---------------- host orchestration ----------------
extern "C" void kernel_launch(void* const* d_in, const int* in_sizes, int n_in,
                              void* d_out, int out_size, void* d_ws, size_t ws_size,
                              hipStream_t stream)
{
  const float* features = (const float*)d_in[0];
  const int*   src      = (const int*)d_in[1];
  const int*   dst      = (const int*)d_in[2];
  const float* W1   = (const float*)d_in[3];
  const float* al1  = (const float*)d_in[4];
  const float* ar1  = (const float*)d_in[5];
  const float* W2   = (const float*)d_in[6];
  const float* al2  = (const float*)d_in[7];
  const float* ar2  = (const float*)d_in[8];
  const float* gn1a = (const float*)d_in[9];
  const float* gn1g = (const float*)d_in[10];
  const float* gn1b = (const float*)d_in[11];
  const float* gn2a = (const float*)d_in[12];
  const float* gn2g = (const float*)d_in[13];
  const float* gn2b = (const float*)d_in[14];
  const float* fc1w = (const float*)d_in[15];
  const float* fc1b = (const float*)d_in[16];
  const float* fc2w = (const float*)d_in[17];
  const float* fc2b = (const float*)d_in[18];
  float* out = (float*)d_out;
  (void)in_sizes; (void)n_in; (void)out_size; (void)ws_size;

  // bump allocator over d_ws (peak ~95 MB; buffers reused across layers)
  char* ws = (char*)d_ws; size_t off = 0;
  auto alloc = [&](size_t bytes) -> float* {
    float* p = (float*)(ws + off);
    off += (bytes + 255) & ~(size_t)255;
    return p;
  };
  float* f1   = alloc((size_t)NTOT * 256 * 4);     // layer1 features / reused as f2
  float* h1   = alloc((size_t)NTOT * 256 * 4);     // layer1 output  / reused as h2
  float* ebuf = alloc((size_t)NEDGE * 4 * 4);      // per-edge e then p
  float* el   = alloc((size_t)NTOT * 4 * 4);
  float* er   = alloc((size_t)NTOT * 4 * 4);
  float* mbuf = alloc((size_t)NTOT * 4 * 4);
  float* zbuf = alloc((size_t)NTOT * 4 * 4);
  float* xm   = alloc((size_t)NTOT * DHID * 4);
  float* xs   = alloc((size_t)NTOT * DHID * 4);
  float* keyb = alloc((size_t)NTOT * 4);
  float* rbuf = alloc((size_t)BGR * 4096 * 4);
  float* hid  = alloc((size_t)BGR * 2048 * 4);

  auto gemm = [&](const float* A, const float* Bw, const float* bias, float* C,
                  int M, int Kd, int P, int relu) {
    if (P % 64 == 0) {
      int tiles = (M / 16) * (P / 64);             // 16x64 tile per wave
      k_wmma_gemm<4><<<(tiles + 7) / 8, 256, 0, stream>>>(A, Bw, bias, C, M, Kd, P, relu);
    } else {
      int tiles = (M / 16) * (P / 16);             // 16x16 tile per wave
      k_wmma_gemm<1><<<(tiles + 7) / 8, 256, 0, stream>>>(A, Bw, bias, C, M, Kd, P, relu);
    }
  };
  auto fill = [&](float* p, float v, size_t n) {
    k_fill<<<(unsigned)((n + 255) / 256), 256, 0, stream>>>(p, v, n);
  };
  auto edge_grid = [&](int H) { return (unsigned)(((long long)NEDGE * H + 255) / 256); };

  // ---------------- layer 1 (H = 4) ----------------
  const int H1 = 4;
  gemm(features, W1, nullptr, f1, NTOT, 64, H1 * DHID, 0);
  k_attn_scores<<<(NTOT * H1 + 255) / 256, 256, 0, stream>>>(f1, al1, ar1, el, er, H1);
  fill(mbuf, -3.4e38f, (size_t)NTOT * H1);
  fill(zbuf, 0.f, (size_t)NTOT * H1);
  fill(h1, 0.f, (size_t)NTOT * H1 * DHID);
  k_edge_score<<<edge_grid(H1), 256, 0, stream>>>(src, dst, el, er, ebuf, mbuf, H1);
  k_edge_exp  <<<edge_grid(H1), 256, 0, stream>>>(dst, ebuf, mbuf, zbuf, H1);
  k_edge_aggr <<<edge_grid(H1), 256, 0, stream>>>(src, dst, ebuf, zbuf, f1, h1, H1);
  k_graphnorm_lrelu<<<BGR * H1, 256, 0, stream>>>(h1, gn1a, gn1g, gn1b, H1);
  k_headmean<<<(NTOT * DHID + 255) / 256, 256, 0, stream>>>(h1, xm, H1);
  k_sortnode<<<(NTOT + 255) / 256, 256, 0, stream>>>(xm, xs, keyb);
  k_toppool<<<BGR, 256, 0, stream>>>(xs, keyb, rbuf, 0);

  // ---------------- layer 2 (H = 2) ----------------
  const int H2 = 2;
  gemm(h1, W2, nullptr, f1, NTOT, 256, H2 * DHID, 0);       // f2 reuses f1
  k_attn_scores<<<(NTOT * H2 + 255) / 256, 256, 0, stream>>>(f1, al2, ar2, el, er, H2);
  fill(mbuf, -3.4e38f, (size_t)NTOT * H2);
  fill(zbuf, 0.f, (size_t)NTOT * H2);
  fill(h1, 0.f, (size_t)NTOT * H2 * DHID);                  // h2 reuses h1
  k_edge_score<<<edge_grid(H2), 256, 0, stream>>>(src, dst, el, er, ebuf, mbuf, H2);
  k_edge_exp  <<<edge_grid(H2), 256, 0, stream>>>(dst, ebuf, mbuf, zbuf, H2);
  k_edge_aggr <<<edge_grid(H2), 256, 0, stream>>>(src, dst, ebuf, zbuf, f1, h1, H2);
  k_graphnorm_lrelu<<<BGR * H2, 256, 0, stream>>>(h1, gn2a, gn2g, gn2b, H2);
  k_headmean<<<(NTOT * DHID + 255) / 256, 256, 0, stream>>>(h1, xm, H2);
  k_sortnode<<<(NTOT + 255) / 256, 256, 0, stream>>>(xm, xs, keyb);
  k_toppool<<<BGR, 256, 0, stream>>>(xs, keyb, rbuf, KTOP * DHID);

  // ---------------- readout MLP ----------------
  gemm(rbuf, fc1w, fc1b, hid, BGR, 2 * KTOP * DHID, KTOP * DHID, 1);  // 32x4096x2048 + relu
  gemm(hid, fc2w, fc2b, out, BGR, KTOP * DHID, 16, 0);                // 32x2048x16
}
